// LaSSMDecoder_22393959481423
// MI455X (gfx1250) — compile-verified, matrix operand order checked by reading, not run on hardware
//
#include <hip/hip_runtime.h>
#include <math.h>

typedef __attribute__((ext_vector_type(16))) _Float16 v16h;
typedef __attribute__((ext_vector_type(8)))  float    v8f;

#define Dm     256
#define Kn     8
#define Qn     2048
#define Nn     32768
#define Hh     8
#define HDd    64
#define STs    64
#define DIN    512
#define CONVC  640
#define NPROJ  1160
#define NPROJP 1184
#define HID    1024
#define KC     4

// ---------------------------------------------------------------------------
// helpers
// ---------------------------------------------------------------------------
static __device__ __forceinline__ float wave_sum32(float s) {
#pragma unroll
  for (int m = 16; m >= 1; m >>= 1) s += __shfl_xor(s, m, 32);
  return s;
}
static __device__ __forceinline__ float siluf(float x) { return x / (1.f + expf(-x)); }
static __device__ __forceinline__ float softplusf(float x) {
  return (x > 20.f) ? x : log1pf(expf(x));
}

// load one 16x32 f16 fragment (A or B operand) from a row-major matrix.
// lane&15 = row/col within tile, lane>>4 selects K offset {0,8}; halfs
// [0..7] come from k..k+7 and halfs [8..15] from k+16..k+23 per ISA layout.
static __device__ __forceinline__ v16h ldfrag(const _Float16* p) {
  union { uint4 q[2]; v16h v; } u;
  u.q[0] = *(const uint4*)p;
  u.q[1] = *(const uint4*)(p + 16);
  return u.v;
}
static __device__ __forceinline__ v8f zero8() {
  v8f z = {0.f, 0.f, 0.f, 0.f, 0.f, 0.f, 0.f, 0.f};
  return z;
}
#define WMMA_F16(a, b, c) \
  __builtin_amdgcn_wmma_f32_16x16x32_f16(false, (a), false, (b), (short)0, (c), false, false)

// ---------------------------------------------------------------------------
// generic C = A * B^T  (A: MxK f16 row-major, B: NxK f16 row-major, C: MxN f32)
// one wave per 32x32 C tile; grid = (N/32, M/32); all dims multiples of 32.
// ---------------------------------------------------------------------------
__global__ void gemm_f16_wmma(const _Float16* __restrict__ A, int lda,
                              const _Float16* __restrict__ B, int ldb,
                              float* __restrict__ C, int ldc, int Kdim) {
  const int n0 = blockIdx.x * 32;
  const int m0 = blockIdx.y * 32;
  const int lane = threadIdx.x;
  const int rA = lane & 15;
  const int kh = (lane >> 4) << 3;

  const _Float16* a0 = A + (size_t)(m0 + rA) * lda + kh;
  const _Float16* a1 = a0 + (size_t)16 * lda;
  const _Float16* b0 = B + (size_t)(n0 + rA) * ldb + kh;
  const _Float16* b1 = b0 + (size_t)16 * ldb;

  v8f c00 = zero8(), c01 = zero8(), c10 = zero8(), c11 = zero8();
  for (int k = 0; k < Kdim; k += 32) {
    v16h fa0 = ldfrag(a0 + k);
    v16h fa1 = ldfrag(a1 + k);
    v16h fb0 = ldfrag(b0 + k);
    v16h fb1 = ldfrag(b1 + k);
    c00 = WMMA_F16(fa0, fb0, c00);
    c01 = WMMA_F16(fa0, fb1, c01);
    c10 = WMMA_F16(fa1, fb0, c10);
    c11 = WMMA_F16(fa1, fb1, c11);
  }
  const int col = n0 + (lane & 15);
  const int ro = (lane >> 4) << 3;  // 0 or 8
#pragma unroll
  for (int r = 0; r < 8; ++r) {
    C[(size_t)(m0 + r + ro) * ldc + col]           = c00[r];
    C[(size_t)(m0 + r + ro) * ldc + col + 16]      = c01[r];
    C[(size_t)(m0 + 16 + r + ro) * ldc + col]      = c10[r];
    C[(size_t)(m0 + 16 + r + ro) * ldc + col + 16] = c11[r];
  }
}

// ---------------------------------------------------------------------------
// conversions
// ---------------------------------------------------------------------------
__global__ void cvt_f16_kernel(const float* __restrict__ s, _Float16* __restrict__ d, int n) {
  int i = blockIdx.x * blockDim.x + threadIdx.x;
  if (i < n) d[i] = (_Float16)s[i];
}
__global__ void fill0_f16_kernel(_Float16* d, int n) {
  int i = blockIdx.x * blockDim.x + threadIdx.x;
  if (i < n) d[i] = (_Float16)0.f;
}

// ---------------------------------------------------------------------------
// KNN top-8: one wave per query
// ---------------------------------------------------------------------------
__global__ void knn_kernel(const float* __restrict__ qpos, const float* __restrict__ spos,
                           int* __restrict__ idx) {
  const int q = blockIdx.x, lane = threadIdx.x;
  const float qx = qpos[q * 3 + 0], qy = qpos[q * 3 + 1], qz = qpos[q * 3 + 2];
  float bd[8];
  int   bi[8];
#pragma unroll
  for (int k = 0; k < 8; ++k) { bd[k] = 1e30f; bi[k] = 0; }
  for (int i = lane; i < Nn; i += 32) {
    float dx = qx - spos[i * 3 + 0];
    float dy = qy - spos[i * 3 + 1];
    float dz = qz - spos[i * 3 + 2];
    float d = dx * dx + dy * dy + dz * dz;
    if (d < bd[7]) {
      int p = 7;
      while (p > 0 && d < bd[p - 1]) { bd[p] = bd[p - 1]; bi[p] = bi[p - 1]; --p; }
      bd[p] = d; bi[p] = i;
    }
  }
  __shared__ float sd[256];
  __shared__ int   si[256];
#pragma unroll
  for (int k = 0; k < 8; ++k) { sd[lane * 8 + k] = bd[k]; si[lane * 8 + k] = bi[k]; }
  __syncthreads();
  if (lane == 0) {
    for (int sel = 0; sel < 8; ++sel) {
      float best = 1e30f; int bj = 0;
      for (int j = 0; j < 256; ++j)
        if (sd[j] < best) { best = sd[j]; bj = j; }
      idx[q * 8 + sel] = si[bj];
      sd[bj] = 1e30f;
    }
  }
}

// ---------------------------------------------------------------------------
// kw = softmax(q @ w_k^T + w_b); one wave per query
// ---------------------------------------------------------------------------
__global__ void kwsm_kernel(const float* __restrict__ qmat, const float* __restrict__ wk,
                            const float* __restrict__ wb, float* __restrict__ kw) {
  const int q = blockIdx.x, lane = threadIdx.x;
  float p[8] = {0, 0, 0, 0, 0, 0, 0, 0};
  for (int d = lane; d < Dm; d += 32) {
    float qv = qmat[(size_t)q * Dm + d];
#pragma unroll
    for (int k = 0; k < 8; ++k) p[k] += qv * wk[k * Dm + d];
  }
#pragma unroll
  for (int k = 0; k < 8; ++k) p[k] = wave_sum32(p[k]);
  if (lane == 0) {
    float l[8], mx = -1e30f;
#pragma unroll
    for (int k = 0; k < 8; ++k) { l[k] = p[k] + wb[k]; mx = fmaxf(mx, l[k]); }
    float se = 0.f;
#pragma unroll
    for (int k = 0; k < 8; ++k) { l[k] = expf(l[k] - mx); se += l[k]; }
#pragma unroll
    for (int k = 0; k < 8; ++k) kw[q * 8 + k] = l[k] / se;
  }
}

// fbar[q,:] = sum_k kw[q,k] * inst_feats[idx[q,k],:]  (f16 out)
__global__ void fbar_kernel(const int* __restrict__ idx, const float* __restrict__ kw,
                            const float* __restrict__ feats, _Float16* __restrict__ out) {
  const int q = blockIdx.x, d = threadIdx.x;
  float acc = 0.f;
#pragma unroll
  for (int k = 0; k < 8; ++k)
    acc += kw[q * 8 + k] * feats[(size_t)idx[q * 8 + k] * Dm + d];
  out[(size_t)q * Dm + d] = (_Float16)acc;
}

// t = q .* vbar (f16 out)
__global__ void tmul_kernel(const float* __restrict__ a, const float* __restrict__ b,
                            _Float16* __restrict__ d, int n) {
  int i = blockIdx.x * blockDim.x + threadIdx.x;
  if (i < n) d[i] = (_Float16)(a[i] * b[i]);
}

// ---------------------------------------------------------------------------
// LayerNorm family (rows of 256, one wave per row, 8 elems/lane)
// ---------------------------------------------------------------------------
__global__ void ln_add_kernel(const float* __restrict__ a, const float* __restrict__ b,
                              float* __restrict__ out) {
  const int row = blockIdx.x, lane = threadIdx.x;
  float v[8]; float s = 0.f;
#pragma unroll
  for (int j = 0; j < 8; ++j) {
    int c = lane + j * 32;
    v[j] = a[(size_t)row * Dm + c] + b[(size_t)row * Dm + c];
    s += v[j];
  }
  float mean = wave_sum32(s) * (1.f / Dm);
  float vs = 0.f;
#pragma unroll
  for (int j = 0; j < 8; ++j) { float dd = v[j] - mean; vs += dd * dd; }
  float r = rsqrtf(wave_sum32(vs) * (1.f / Dm) + 1e-5f);
#pragma unroll
  for (int j = 0; j < 8; ++j)
    out[(size_t)row * Dm + lane + j * 32] = (v[j] - mean) * r;
}

// xs[b,i,:] = LN(x[order[b,i],:]) in f16; grid (Qn, 2)
__global__ void ln_perm_kernel(const float* __restrict__ x, const int* __restrict__ order,
                               _Float16* __restrict__ xs) {
  const int i = blockIdx.x, b = blockIdx.y, lane = threadIdx.x;
  const int src = order[b * Qn + i];
  float v[8]; float s = 0.f;
#pragma unroll
  for (int j = 0; j < 8; ++j) { v[j] = x[(size_t)src * Dm + lane + j * 32]; s += v[j]; }
  float mean = wave_sum32(s) * (1.f / Dm);
  float vs = 0.f;
#pragma unroll
  for (int j = 0; j < 8; ++j) { float dd = v[j] - mean; vs += dd * dd; }
  float r = rsqrtf(wave_sum32(vs) * (1.f / Dm) + 1e-5f);
  _Float16* dst = xs + ((size_t)b * Qn + i) * Dm;
#pragma unroll
  for (int j = 0; j < 8; ++j) dst[lane + j * 32] = (_Float16)((v[j] - mean) * r);
}

// x = LN(x + 0.5*(yout[b0 path] + yout[b1 path])) (in place per row)
__global__ void ln_unsort_kernel(float* __restrict__ x, const float* __restrict__ yout,
                                 const int* __restrict__ inv) {
  const int row = blockIdx.x, lane = threadIdx.x;
  const int i0 = inv[row], i1 = inv[Qn + row];
  float v[8]; float s = 0.f;
#pragma unroll
  for (int j = 0; j < 8; ++j) {
    int c = lane + j * 32;
    float add = 0.5f * (yout[(size_t)i0 * Dm + c] + yout[((size_t)Qn + i1) * Dm + c]);
    v[j] = x[(size_t)row * Dm + c] + add;
    s += v[j];
  }
  float mean = wave_sum32(s) * (1.f / Dm);
  float vs = 0.f;
#pragma unroll
  for (int j = 0; j < 8; ++j) { float dd = v[j] - mean; vs += dd * dd; }
  float r = rsqrtf(wave_sum32(vs) * (1.f / Dm) + 1e-5f);
#pragma unroll
  for (int j = 0; j < 8; ++j)
    x[(size_t)row * Dm + lane + j * 32] = (v[j] - mean) * r;
}

__global__ void ln_final_kernel(const float* __restrict__ x, const float* __restrict__ h2,
                                const float* __restrict__ b2, float* __restrict__ out) {
  const int row = blockIdx.x, lane = threadIdx.x;
  float v[8]; float s = 0.f;
#pragma unroll
  for (int j = 0; j < 8; ++j) {
    int c = lane + j * 32;
    v[j] = x[(size_t)row * Dm + c] + h2[(size_t)row * Dm + c] + b2[c];
    s += v[j];
  }
  float mean = wave_sum32(s) * (1.f / Dm);
  float vs = 0.f;
#pragma unroll
  for (int j = 0; j < 8; ++j) { float dd = v[j] - mean; vs += dd * dd; }
  float r = rsqrtf(wave_sum32(vs) * (1.f / Dm) + 1e-5f);
#pragma unroll
  for (int j = 0; j < 8; ++j)
    out[(size_t)row * Dm + lane + j * 32] = (v[j] - mean) * r;
}

// ---------------------------------------------------------------------------
// SSM pieces
// ---------------------------------------------------------------------------
// causal depthwise conv over xBC slice of proj + silu; one block per (b,t) row
__global__ void convsilu_kernel(const float* __restrict__ proj, const float* __restrict__ wc,
                                const float* __restrict__ bc, float* __restrict__ out) {
  const int row = blockIdx.x;            // 0..4095
  const int b = row >> 11, t = row & 2047;
  for (int c = threadIdx.x; c < CONVC; c += blockDim.x) {
    float acc = bc[c];
#pragma unroll
    for (int j = 0; j < KC; ++j) {
      int tt = t - (KC - 1) + j;
      if (tt >= 0)
        acc += proj[((size_t)(b * Qn + tt)) * NPROJP + DIN + c] * wc[c * KC + j];
    }
    out[(size_t)row * CONVC + c] = siluf(acc);
  }
}

__global__ void dt_kernel(const float* __restrict__ proj, const float* __restrict__ dtb,
                          float* __restrict__ dt, int n) {
  int i = blockIdx.x * blockDim.x + threadIdx.x;
  if (i < n) {
    int row = i >> 3, h = i & 7;
    dt[i] = softplusf(proj[(size_t)row * NPROJP + DIN + CONVC + h] + dtb[h]);
  }
}

// sequential recurrence: grid 16 = (b,h); 64 threads, each owns one d-row,
// 64-wide state kept in registers; B/C staged through LDS each step.
__global__ void scan_kernel(const float* __restrict__ conv, const float* __restrict__ dt,
                            const float* __restrict__ Alog, const float* __restrict__ Dv,
                            float* __restrict__ ybuf) {
  const int bb = blockIdx.x >> 3, hh = blockIdx.x & 7;
  const int d = threadIdx.x;  // 0..63
  const float A  = -expf(Alog[hh]);
  const float Dh = Dv[hh];
  float st[STs];
#pragma unroll
  for (int s = 0; s < STs; ++s) st[s] = 0.f;
  __shared__ float Bs[STs], Cs[STs];
  for (int t = 0; t < Qn; ++t) {
    const size_t row = (size_t)bb * Qn + t;
    Bs[d] = conv[row * CONVC + DIN + d];
    Cs[d] = conv[row * CONVC + DIN + STs + d];
    const float dtv = dt[row * Hh + hh];
    const float dA = expf(dtv * A);
    const float xv = conv[row * CONVC + hh * HDd + d];
    __syncthreads();
    const float dx = dtv * xv;
    float y = 0.f;
#pragma unroll
    for (int s = 0; s < STs; ++s) {
      st[s] = st[s] * dA + Bs[s] * dx;
      y += st[s] * Cs[s];
    }
    ybuf[row * DIN + hh * HDd + d] = y + Dh * xv;
    __syncthreads();
  }
}

// g = rmsnorm(y * silu(z)) * rmsw  (f16 out); one wave per row of 512
__global__ void gatenorm_kernel(const float* __restrict__ ybuf, const float* __restrict__ proj,
                                const float* __restrict__ rmsw, _Float16* __restrict__ g) {
  const int row = blockIdx.x, lane = threadIdx.x;
  float v[16]; float ss = 0.f;
#pragma unroll
  for (int j = 0; j < 16; ++j) {
    int c = lane + j * 32;
    float z = proj[(size_t)row * NPROJP + c];
    float y = ybuf[(size_t)row * DIN + c] * siluf(z);
    v[j] = y; ss += y * y;
  }
  float r = rsqrtf(wave_sum32(ss) * (1.f / DIN) + 1e-5f);
#pragma unroll
  for (int j = 0; j < 16; ++j) {
    int c = lane + j * 32;
    g[(size_t)row * DIN + c] = (_Float16)(v[j] * r * rmsw[c]);
  }
}

__global__ void inv_kernel(const int* __restrict__ order, int* __restrict__ inv, int n) {
  int i = blockIdx.x * blockDim.x + threadIdx.x;
  if (i < n) {
    int b = i >> 11, pos = i & 2047;
    inv[b * Qn + order[i]] = pos;
  }
}

__global__ void gelu_kernel(const float* __restrict__ h1, const float* __restrict__ b1,
                            _Float16* __restrict__ g, int n) {
  int i = blockIdx.x * blockDim.x + threadIdx.x;
  if (i < n) {
    float v = h1[i] + b1[i % HID];
    g[i] = (_Float16)(0.5f * v * (1.f + erff(v * 0.70710678118654752f)));
  }
}

// ---------------------------------------------------------------------------
// orchestration
// ---------------------------------------------------------------------------
extern "C" void kernel_launch(void* const* d_in, const int* in_sizes, int n_in,
                              void* d_out, int out_size, void* d_ws, size_t ws_size,
                              hipStream_t stream) {
  const float* query      = (const float*)d_in[0];
  const float* query_pos  = (const float*)d_in[1];
  const float* inst_feats = (const float*)d_in[2];
  const float* sp_coords  = (const float*)d_in[3];
  const float* w_q        = (const float*)d_in[4];
  const float* w_v        = (const float*)d_in[5];
  const float* w_o        = (const float*)d_in[6];
  const float* w_k        = (const float*)d_in[7];
  const float* w_b        = (const float*)d_in[8];
  const float* ssm_Win    = (const float*)d_in[9];
  const float* ssm_Wconv  = (const float*)d_in[10];
  const float* ssm_bconv  = (const float*)d_in[11];
  const float* ssm_Alog   = (const float*)d_in[12];
  const float* ssm_D      = (const float*)d_in[13];
  const float* ssm_dtb    = (const float*)d_in[14];
  const float* ssm_rmsw   = (const float*)d_in[15];
  const float* ssm_Wout   = (const float*)d_in[16];
  const float* ffn_w1     = (const float*)d_in[17];
  const float* ffn_b1     = (const float*)d_in[18];
  const float* ffn_w2     = (const float*)d_in[19];
  const float* ffn_b2     = (const float*)d_in[20];
  const int*   order      = (const int*)d_in[21];
  float* out = (float*)d_out;

  char* wp = (char*)d_ws;
  auto alloc = [&](size_t bytes) -> void* {
    void* r = (void*)wp;
    wp += (bytes + 255) & ~(size_t)255;
    return r;
  };
  _Float16* qryF16  = (_Float16*)alloc((size_t)Qn * Dm * 2);
  _Float16* wqF16   = (_Float16*)alloc((size_t)Dm * Dm * 2);
  _Float16* wvF16   = (_Float16*)alloc((size_t)Dm * Dm * 2);
  _Float16* woF16   = (_Float16*)alloc((size_t)Dm * Dm * 2);
  _Float16* winF16  = (_Float16*)alloc((size_t)2 * NPROJP * Dm * 2);
  _Float16* woutF16 = (_Float16*)alloc((size_t)2 * Dm * DIN * 2);
  _Float16* w1F16   = (_Float16*)alloc((size_t)HID * Dm * 2);
  _Float16* w2F16   = (_Float16*)alloc((size_t)Dm * HID * 2);
  int*      idx     = (int*)alloc((size_t)Qn * Kn * 4);
  float*    qmat    = (float*)alloc((size_t)Qn * Dm * 4);
  float*    kw      = (float*)alloc((size_t)Qn * Kn * 4);
  _Float16* fbarF16 = (_Float16*)alloc((size_t)Qn * Dm * 2);
  float*    vbar    = (float*)alloc((size_t)Qn * Dm * 4);
  _Float16* tF16    = (_Float16*)alloc((size_t)Qn * Dm * 2);
  float*    obuf    = (float*)alloc((size_t)Qn * Dm * 4);
  float*    xbuf    = (float*)alloc((size_t)Qn * Dm * 4);
  _Float16* xsF16   = (_Float16*)alloc((size_t)2 * Qn * Dm * 2);
  float*    proj    = (float*)alloc((size_t)2 * Qn * NPROJP * 4);
  float*    convout = (float*)alloc((size_t)2 * Qn * CONVC * 4);
  float*    dtbuf   = (float*)alloc((size_t)2 * Qn * Hh * 4);
  float*    ybuf    = (float*)alloc((size_t)2 * Qn * DIN * 4);
  _Float16* gF16    = (_Float16*)alloc((size_t)2 * Qn * DIN * 2);
  float*    yout    = (float*)alloc((size_t)2 * Qn * Dm * 4);
  int*      inv     = (int*)alloc((size_t)2 * Qn * 4);
  _Float16* xcF16   = (_Float16*)alloc((size_t)Qn * Dm * 2);
  float*    h1buf   = (float*)alloc((size_t)Qn * HID * 4);
  _Float16* geF16   = (_Float16*)alloc((size_t)Qn * HID * 2);
  float*    h2buf   = (float*)alloc((size_t)Qn * Dm * 4);

  auto cvt = [&](const float* s, _Float16* d, int n) {
    cvt_f16_kernel<<<(n + 255) / 256, 256, 0, stream>>>(s, d, n);
  };

  // --- weight / input conversions to f16 ---
  cvt(query, qryF16, Qn * Dm);
  cvt(w_q, wqF16, Dm * Dm);
  cvt(w_v, wvF16, Dm * Dm);
  cvt(w_o, woF16, Dm * Dm);
  fill0_f16_kernel<<<(2 * NPROJP * Dm + 255) / 256, 256, 0, stream>>>(winF16, 2 * NPROJP * Dm);
  cvt(ssm_Win,                 winF16,                 NPROJ * Dm);
  cvt(ssm_Win + NPROJ * Dm,    winF16 + NPROJP * Dm,   NPROJ * Dm);
  cvt(ssm_Wout, woutF16, 2 * Dm * DIN);
  cvt(ffn_w1, w1F16, HID * Dm);
  cvt(ffn_w2, w2F16, Dm * HID);
  inv_kernel<<<(2 * Qn + 255) / 256, 256, 0, stream>>>(order, inv, 2 * Qn);

  // --- KNN aggregation ---
  knn_kernel<<<Qn, 32, 0, stream>>>(query_pos, sp_coords, idx);
  gemm_f16_wmma<<<dim3(Dm / 32, Qn / 32), 32, 0, stream>>>(qryF16, Dm, wqF16, Dm, qmat, Dm, Dm);
  kwsm_kernel<<<Qn, 32, 0, stream>>>(qmat, w_k, w_b, kw);
  fbar_kernel<<<Qn, Dm, 0, stream>>>(idx, kw, inst_feats, fbarF16);
  gemm_f16_wmma<<<dim3(Dm / 32, Qn / 32), 32, 0, stream>>>(fbarF16, Dm, wvF16, Dm, vbar, Dm, Dm);
  tmul_kernel<<<(Qn * Dm + 255) / 256, 256, 0, stream>>>(qmat, vbar, tF16, Qn * Dm);
  gemm_f16_wmma<<<dim3(Dm / 32, Qn / 32), 32, 0, stream>>>(tF16, Dm, woF16, Dm, obuf, Dm, Dm);
  ln_add_kernel<<<Qn, 32, 0, stream>>>(obuf, query, xbuf);

  // --- 2 SSM layers ---
  for (int l = 0; l < 2; ++l) {
    ln_perm_kernel<<<dim3(Qn, 2), 32, 0, stream>>>(xbuf, order, xsF16);
    gemm_f16_wmma<<<dim3(NPROJP / 32, (2 * Qn) / 32), 32, 0, stream>>>(
        xsF16, Dm, winF16 + (size_t)l * NPROJP * Dm, Dm, proj, NPROJP, Dm);
    convsilu_kernel<<<2 * Qn, 256, 0, stream>>>(
        proj, ssm_Wconv + (size_t)l * CONVC * KC, ssm_bconv + (size_t)l * CONVC, convout);
    dt_kernel<<<(2 * Qn * Hh + 255) / 256, 256, 0, stream>>>(
        proj, ssm_dtb + l * Hh, dtbuf, 2 * Qn * Hh);
    scan_kernel<<<16, 64, 0, stream>>>(convout, dtbuf, ssm_Alog + l * Hh, ssm_D + l * Hh, ybuf);
    gatenorm_kernel<<<2 * Qn, 32, 0, stream>>>(ybuf, proj, ssm_rmsw + (size_t)l * DIN, gF16);
    gemm_f16_wmma<<<dim3(Dm / 32, (2 * Qn) / 32), 32, 0, stream>>>(
        gF16, DIN, woutF16 + (size_t)l * Dm * DIN, DIN, yout, Dm, DIN);
    ln_unsort_kernel<<<Qn, 32, 0, stream>>>(xbuf, yout, inv);
  }

  // --- FFN ---
  cvt(xbuf, xcF16, Qn * Dm);
  gemm_f16_wmma<<<dim3(HID / 32, Qn / 32), 32, 0, stream>>>(xcF16, Dm, w1F16, Dm, h1buf, HID, Dm);
  gelu_kernel<<<(Qn * HID + 255) / 256, 256, 0, stream>>>(h1buf, ffn_b1, geF16, Qn * HID);
  gemm_f16_wmma<<<dim3(Dm / 32, Qn / 32), 32, 0, stream>>>(geF16, HID, w2F16, HID, h2buf, Dm, HID);
  ln_final_kernel<<<Qn, 32, 0, stream>>>(xbuf, h2buf, ffn_b2, out);
}
